// GCNModel_60601988547217
// MI455X (gfx1250) — matrix operand order repaired and checked by reading, not compiled, not run on hardware
//
#include <hip/hip_runtime.h>
#include <hip/hip_bf16.h>

// ---------------------------------------------------------------------------
// GCN + question MLP for MI455X (gfx1250, wave32, WMMA).
// Dense matmuls -> v_wmma_f32_16x16x32_bf16 (f32 accum). Irregular graph
// aggregation -> coalesced f32 global atomics (L2-resident: each N x 128 f32
// activation is 51 MB < 192 MB L2).
// ---------------------------------------------------------------------------

#define N_NODES 100000
#define N_EDGES 1600000
#define N_B     64

typedef __attribute__((ext_vector_type(16))) __bf16 v16bf;
typedef __attribute__((ext_vector_type(8)))  float  v8f;

// ---------------------------- utility kernels ------------------------------

__global__ void zero_kernel(float* __restrict__ p, int n) {
  int i = blockIdx.x * blockDim.x + threadIdx.x;
  if (i < n) p[i] = 0.0f;
}

__global__ void cvt_bf16_kernel(const float* __restrict__ in,
                                __bf16* __restrict__ out, int n) {
  int i = blockIdx.x * blockDim.x + threadIdx.x;
  if (i < n) out[i] = (__bf16)in[i];
}

__global__ void deg_kernel(const long long* __restrict__ dst,
                           float* __restrict__ deg, int E) {
  int i = blockIdx.x * blockDim.x + threadIdx.x;
  if (i < E) atomicAdd(&deg[(int)dst[i]], 1.0f);
}

__global__ void dis_kernel(const float* __restrict__ deg,
                           float* __restrict__ dis, int n) {
  int i = blockIdx.x * blockDim.x + threadIdx.x;
  if (i < n) dis[i] = rsqrtf(1.0f + deg[i]);
}

// agg[dst] += tmp[src] * dis[src]*dis[dst]; one thread handles 4 dims of one
// edge so reads are coalesced 512B rows and atomics land in L2.
__global__ void scatter_kernel(const long long* __restrict__ src,
                               const long long* __restrict__ dst,
                               const float* __restrict__ dis,
                               const float* __restrict__ tmp,
                               float* __restrict__ agg, int E) {
  int i = blockIdx.x * blockDim.x + threadIdx.x;
  if (i >= E * 32) return;
  int e = i >> 5;
  int d = (i & 31) << 2;
  int s = (int)src[e];
  int t = (int)dst[e];
  float norm = dis[s] * dis[t];
  float4 v = *(const float4*)(tmp + (size_t)s * 128 + d);
  float* base = agg + (size_t)t * 128 + d;
  atomicAdd(base + 0, v.x * norm);
  atomicAdd(base + 1, v.y * norm);
  atomicAdd(base + 2, v.z * norm);
  atomicAdd(base + 3, v.w * norm);
}

// h = elu(agg + tmp*dis^2 + b) [+ residual]
__global__ void gcn_epilogue_kernel(const float* __restrict__ agg,
                                    const float* __restrict__ tmp,
                                    const float* __restrict__ dis,
                                    const float* __restrict__ bias,
                                    const float* __restrict__ res,
                                    float* __restrict__ out, int n) {
  int i = blockIdx.x * blockDim.x + threadIdx.x;
  if (i >= n) return;
  int row = i >> 7;
  int d   = i & 127;
  float ds = dis[row];
  float v = agg[i] + tmp[i] * ds * ds + bias[d];
  v = (v > 0.0f) ? v : (expf(v) - 1.0f);
  if (res) v += res[i];
  out[i] = v;
}

// combined = [h | q[batch]]
__global__ void combine_kernel(const float* __restrict__ h,
                               const float* __restrict__ q,
                               const long long* __restrict__ batch,
                               float* __restrict__ comb, int n) {
  int i = blockIdx.x * blockDim.x + threadIdx.x;
  if (i >= n) return;
  int row = i >> 8;
  int d   = i & 255;
  comb[i] = (d < 128) ? h[(size_t)row * 128 + d]
                      : q[(size_t)((int)batch[row]) * 128 + (d - 128)];
}

// ------------------------------ WMMA GEMM ----------------------------------
// out[M x 128] = A[M x K] @ Wbf16[K x 128]  (+bias, optional ELU)
// 256 threads = 8 waves; each wave owns a 16-row tile and all 8 column tiles.
// A fragment (16-bit A 16x32 ISA layout): lane L holds row M=L&15,
//   K runs [k0, k0+8) and [k0+16, k0+24) with k0 = (L>>4)*8  -> two float4x2.
// B fragment: lane L holds W column N=L&15, K in the same order as A
//   ([k0, k0+8) and [k0+16, k0+24)) -> strided gather from row-major bf16 weights.
__global__ __launch_bounds__(256) void gemm_wmma_kernel(
    const float* __restrict__ A, const __bf16* __restrict__ W,
    const float* __restrict__ bias, float* __restrict__ out,
    int M, int K, int do_elu) {
  const int lane    = threadIdx.x & 31;
  const int wave    = threadIdx.x >> 5;
  const int tileRow = blockIdx.x * 128 + wave * 16;

  int mRow = tileRow + (lane & 15);
  const int m  = (mRow < M) ? mRow : (M - 1);   // clamp: EXEC must stay all-1s
  const int k0 = (lane >> 4) << 3;

  v8f acc[8];
#pragma unroll
  for (int t = 0; t < 8; ++t) {
    v8f z = {0.f, 0.f, 0.f, 0.f, 0.f, 0.f, 0.f, 0.f};
    acc[t] = z;
  }

  const int K32 = K >> 5;
  for (int c = 0; c < K32; ++c) {
    const float* ar = A + (size_t)m * K + (c << 5) + k0;
    float4 f0 = *(const float4*)(ar);
    float4 f1 = *(const float4*)(ar + 4);
    float4 f2 = *(const float4*)(ar + 16);
    float4 f3 = *(const float4*)(ar + 20);
    v16bf a;
    a[0]  = (__bf16)f0.x; a[1]  = (__bf16)f0.y;
    a[2]  = (__bf16)f0.z; a[3]  = (__bf16)f0.w;
    a[4]  = (__bf16)f1.x; a[5]  = (__bf16)f1.y;
    a[6]  = (__bf16)f1.z; a[7]  = (__bf16)f1.w;
    a[8]  = (__bf16)f2.x; a[9]  = (__bf16)f2.y;
    a[10] = (__bf16)f2.z; a[11] = (__bf16)f2.w;
    a[12] = (__bf16)f3.x; a[13] = (__bf16)f3.y;
    a[14] = (__bf16)f3.z; a[15] = (__bf16)f3.w;

    const __bf16* wr = W + (size_t)((c << 5) + k0) * 128 + (lane & 15);
#pragma unroll
    for (int t = 0; t < 8; ++t) {
      v16bf b;
#pragma unroll
      for (int s = 0; s < 8; ++s) {
        b[s]     = wr[(size_t)s * 128 + t * 16];
        b[s + 8] = wr[(size_t)(s + 16) * 128 + t * 16];
      }
      acc[t] = __builtin_amdgcn_wmma_f32_16x16x32_bf16(
          false, a, false, b, (short)0, acc[t], false, false);
    }
  }

  // C/D layout: VGPR r, lanes 0-15 -> M = r, lanes 16-31 -> M = r + 8.
  const int mBase = tileRow + ((lane >> 4) << 3);
  const int nOff  = lane & 15;
#pragma unroll
  for (int t = 0; t < 8; ++t) {
    int col = t * 16 + nOff;
    float bv = bias ? bias[col] : 0.0f;
#pragma unroll
    for (int r = 0; r < 8; ++r) {
      int row = mBase + r;
      if (row < M) {
        float v = acc[t][r] + bv;
        if (do_elu) v = (v > 0.0f) ? v : (expf(v) - 1.0f);
        out[(size_t)row * 128 + col] = v;
      }
    }
  }
}

// ------------------------------- launch ------------------------------------

static inline unsigned cdiv_ll(long long a, long long b) {
  return (unsigned)((a + b - 1) / b);
}

extern "C" void kernel_launch(void* const* d_in, const int* in_sizes, int n_in,
                              void* d_out, int out_size, void* d_ws,
                              size_t ws_size, hipStream_t stream) {
  (void)in_sizes; (void)n_in; (void)out_size; (void)ws_size;
  const int N = N_NODES, E = N_EDGES;

  const float*     x     = (const float*)d_in[0];
  const long long* ei    = (const long long*)d_in[1];
  const long long* batch = (const long long*)d_in[2];
  const float*     qe    = (const float*)d_in[3];
  const float* w[4] = {(const float*)d_in[4], (const float*)d_in[6],
                       (const float*)d_in[8], (const float*)d_in[10]};
  const float* b[4] = {(const float*)d_in[5], (const float*)d_in[7],
                       (const float*)d_in[9], (const float*)d_in[11]};
  const float* fc0_w = (const float*)d_in[12];
  const float* fc0_b = (const float*)d_in[13];
  const float* fc1_w = (const float*)d_in[14];
  const float* fc1_b = (const float*)d_in[15];
  const float* fc2_w = (const float*)d_in[16];
  const float* fc2_b = (const float*)d_in[17];
  float* outp = (float*)d_out;

  // ---- workspace carve (floats first, bf16 weights after; base is aligned)
  float* fws  = (float*)d_ws;
  float* tmp  = fws;                         // N*128  (h @ w)
  float* agg  = tmp  + (size_t)N * 128;      // N*128  (scatter target)
  float* hA   = agg  + (size_t)N * 128;      // N*128
  float* hRes = hA   + (size_t)N * 128;      // N*128  (residual = h after L2)
  float* hB   = hRes + (size_t)N * 128;      // N*128  (final h)
  float* deg  = hB   + (size_t)N * 128;      // N
  float* dis  = deg  + N;                    // N
  float* q    = dis  + N;                    // B*128
  __bf16* wb0  = (__bf16*)(q + (size_t)N_B * 128);
  __bf16* wb1  = wb0  + 16384;
  __bf16* wb2  = wb1  + 16384;
  __bf16* wb3  = wb2  + 16384;
  __bf16* fwb0 = wb3  + 16384;               // 768*128
  __bf16* fwb1 = fwb0 + 98304;               // 256*128
  __bf16* fwb2 = fwb1 + 32768;               // 128*128
  float* comb = tmp;  // [N x 256] aliases contiguous (tmp, agg) — dead by then

  const long long* srcp = ei;
  const long long* dstp = ei + E;
  dim3 blk(256);

  // degrees with self loops -> deg^-1/2
  zero_kernel<<<cdiv_ll(N, 256), blk, 0, stream>>>(deg, N);
  deg_kernel<<<cdiv_ll(E, 256), blk, 0, stream>>>(dstp, deg, E);
  dis_kernel<<<cdiv_ll(N, 256), blk, 0, stream>>>(deg, dis, N);

  // weights -> bf16 (tiny; stays L2-resident)
  cvt_bf16_kernel<<<cdiv_ll(16384, 256), blk, 0, stream>>>(w[0], wb0, 16384);
  cvt_bf16_kernel<<<cdiv_ll(16384, 256), blk, 0, stream>>>(w[1], wb1, 16384);
  cvt_bf16_kernel<<<cdiv_ll(16384, 256), blk, 0, stream>>>(w[2], wb2, 16384);
  cvt_bf16_kernel<<<cdiv_ll(16384, 256), blk, 0, stream>>>(w[3], wb3, 16384);
  cvt_bf16_kernel<<<cdiv_ll(98304, 256), blk, 0, stream>>>(fc0_w, fwb0, 98304);
  cvt_bf16_kernel<<<cdiv_ll(32768, 256), blk, 0, stream>>>(fc1_w, fwb1, 32768);
  cvt_bf16_kernel<<<cdiv_ll(16384, 256), blk, 0, stream>>>(fc2_w, fwb2, 16384);

  // q = elu(qe @ fc0_w + fc0_b)   [64 x 768] @ [768 x 128]
  gemm_wmma_kernel<<<cdiv_ll(N_B, 128), blk, 0, stream>>>(
      qe, fwb0, fc0_b, q, N_B, 768, 1);

  // 4 GCN layers
  const float*   hin      = x;
  float*         houts[4] = {hA, hRes, hA, hB};
  const float*   resv[4]  = {nullptr, nullptr, nullptr, hRes};
  const __bf16*  wbs[4]   = {wb0, wb1, wb2, wb3};
  for (int L = 0; L < 4; ++L) {
    gemm_wmma_kernel<<<cdiv_ll(N, 128), blk, 0, stream>>>(
        hin, wbs[L], nullptr, tmp, N, 128, 0);
    zero_kernel<<<cdiv_ll((long long)N * 128, 256), blk, 0, stream>>>(
        agg, N * 128);
    scatter_kernel<<<cdiv_ll((long long)E * 32, 256), blk, 0, stream>>>(
        srcp, dstp, dis, tmp, agg, E);
    gcn_epilogue_kernel<<<cdiv_ll((long long)N * 128, 256), blk, 0, stream>>>(
        agg, tmp, dis, b[L], resv[L], houts[L], N * 128);
    hin = houts[L];
  }

  // combined = [hB | q[batch]]  -> fc1 (elu) -> fc2
  combine_kernel<<<cdiv_ll((long long)N * 256, 256), blk, 0, stream>>>(
      hB, q, batch, comb, N * 256);
  gemm_wmma_kernel<<<cdiv_ll(N, 128), blk, 0, stream>>>(
      comb, fwb1, fc1_b, hA, N, 256, 1);
  gemm_wmma_kernel<<<cdiv_ll(N, 128), blk, 0, stream>>>(
      hA, fwb2, fc2_b, outp, N, 128, 0);
}
